// Edge_Update_Only_16449724744597
// MI455X (gfx1250) — matrix operand order repaired and checked by reading, not compiled, not run on hardware
//
#include <hip/hip_runtime.h>
#include <hip/hip_bf16.h>

typedef __attribute__((ext_vector_type(16))) _Float16 v16h;
typedef __attribute__((ext_vector_type(8)))  _Float16 v8h;
typedef __attribute__((ext_vector_type(2)))  _Float16 v2h;
typedef __attribute__((ext_vector_type(8)))  float    v8f;
typedef __attribute__((ext_vector_type(4)))  float    v4f;

#define NB      4
#define NN      384
#define E_PER_B (NN * NN)          // 147456
#define ROWS    (NB * E_PER_B)     // 589824
#define D_IN    160                // 2*64 + 32
#define D_H     128
#define D_OUT   64

#define W_TOTAL (D_H * D_IN + D_H * D_H + D_OUT * D_H)  // 45056 f16
#define LDS_H_OFF W_TOTAL                               // per-wave h scratch after weights
#define LDS_F16_TOTAL (W_TOTAL + 8 * 32 * D_H)          // 77824 f16 = 155648 B

// ---- weight f32 -> f16 pre-pass ----------------------------------------
__global__ void cvt_weights(const float* __restrict__ Wt,
                            const float* __restrict__ W1,
                            const float* __restrict__ W2,
                            _Float16* __restrict__ o) {
    int t = blockIdx.x * 256 + threadIdx.x;
    if (t < D_H * D_IN)                    o[t] = (_Float16)Wt[t];
    else if (t < D_H * D_IN + D_H * D_H)   o[t] = (_Float16)W1[t - D_H * D_IN];
    else if (t < W_TOTAL)                  o[t] = (_Float16)W2[t - D_H * D_IN - D_H * D_H];
}

// ---- per-(b,i) src-node bias: bias[row][col] = sum_{k<64} node[row][k]*Wt[col][k]
// Only 1536x128 outputs (12.6 MFLOP) -- computed once, reused by all 384 j-rows.
__global__ void src_bias_kernel(const float* __restrict__ node,
                                const float* __restrict__ Wt,
                                float* __restrict__ bias) {
    const int row = blockIdx.x;    // (b*NN + i), 0..1535
    const int col = threadIdx.x;   // 0..127
    const float* s = node + (size_t)row * 64;
    const float* w = Wt + (size_t)col * D_IN;   // src part = first 64 of each Wt row
    float acc = 0.f;
#pragma unroll
    for (int k = 0; k < 64; ++k) acc += s[k] * w[k];
    bias[(size_t)row * D_H + col] = acc;
}

// Build 16-bit A fragment (16x32, MxK): per-lane 8 f16 at K=g0..g0+7 and 8 at
// K=g1..g1+7 (lanes 0-15 vs 16-31 offset handled by caller) from f32 global.
__device__ __forceinline__ v16h make_a_frag_g(const float* p0, const float* p1) {
    v4f x0 = *(const v4f*)p0;
    v4f x1 = *(const v4f*)(p0 + 4);
    v4f x2 = *(const v4f*)p1;
    v4f x3 = *(const v4f*)(p1 + 4);
    v16h r;
#pragma unroll
    for (int t = 0; t < 4; ++t) {
        r[t]      = (_Float16)x0[t];
        r[4 + t]  = (_Float16)x1[t];
        r[8 + t]  = (_Float16)x2[t];
        r[12 + t] = (_Float16)x3[t];
    }
    return r;
}

// Two 16B-aligned LDS loads -> one fragment register set.
__device__ __forceinline__ v16h load_frag_lds(const _Float16* p0, const _Float16* p1) {
    v8h lo = *(const v8h*)p0;
    v8h hi = *(const v8h*)p1;
    v16h r;
#pragma unroll
    for (int t = 0; t < 8; ++t) { r[t] = lo[t]; r[8 + t] = hi[t]; }
    return r;
}

// Packed leaky-relu (slope 0.1) + f16 convert + scatter D-layout rows into sH.
__device__ __forceinline__ void relu_store_h(_Float16* base, const v8f& acc) {
#pragma unroll
    for (int r = 0; r < 8; r += 2) {
        v2h p;
        p[0] = (_Float16)acc[r];
        p[1] = (_Float16)acc[r + 1];                    // -> v_cvt_pk_f16_f32
        v2h q = p * (_Float16)0.1f;                     // -> v_pk_mul_f16
        p = __builtin_elementwise_max(p, q);            // -> v_pk_max_f16
        base[(size_t)r * D_H]       = p[0];
        base[(size_t)(r + 1) * D_H] = p[1];
    }
}

#define WMMA_F16(A, B, C) \
    __builtin_amdgcn_wmma_f32_16x16x32_f16(false, (A), false, (B), (short)0, (C), false, false)

__global__ __launch_bounds__(256) void edge_mlp_kernel(
        const float* __restrict__ node_fts,   // [B, N, 64]
        const float* __restrict__ edge_fts,   // [B, N*N, 32]
        const _Float16* __restrict__ wf16,    // converted weights in d_ws
        const float* __restrict__ bias,       // [B*N, 128] src contribution
        float* __restrict__ out) {            // [B, N*N, 64]
    extern __shared__ _Float16 smem[];
    _Float16* sWt = smem;                               // [128][160]
    _Float16* sW1 = smem + D_H * D_IN;                  // [128][128]
    _Float16* sW2 = smem + D_H * D_IN + D_H * D_H;      // [64][128]

    const int tid  = threadIdx.x;
    const int wave = tid >> 5;
    const int lane = tid & 31;
    const int hlf  = lane >> 4;     // 0: lanes 0-15, 1: lanes 16-31
    const int lr   = lane & 15;

    _Float16* sH = smem + LDS_H_OFF + wave * (32 * D_H);  // per-wave 32x128 h scratch

    // ---- stage all weights into LDS (88 KB), 22 x uint4 per thread ----
    {
        const uint4* gw = (const uint4*)wf16;
        uint4* sw = (uint4*)smem;
#pragma unroll
        for (int t = 0; t < (W_TOTAL * 2 / 16) / 256; ++t)
            sw[t * 256 + tid] = gw[t * 256 + tid];
    }
    __syncthreads();

    // ---- this wave's 32 rows (two 16-row M-tiles): same (b,i), j = j0..j0+31 ----
    const int row0 = blockIdx.x * 256 + wave * 32;
    const int b    = row0 / E_PER_B;
    const int rem  = row0 % E_PER_B;
    const int i    = rem / NN;
    const int j0   = rem % NN;

    const float* biasp = bias + (size_t)(b * NN + i) * D_H;
    const float* dstp[2];
    const float* edgp[2];
#pragma unroll
    for (int m = 0; m < 2; ++m) {
        dstp[m] = node_fts + (size_t)(b * NN + j0 + m * 16 + lr) * 64;
        edgp[m] = edge_fts + (size_t)(b * E_PER_B + i * NN + j0 + m * 16 + lr) * 32;
    }

    // ---- layer-1 A fragments (dst + edge only; src folded into bias) ----
    v16h a1[2][3];
#pragma unroll
    for (int m = 0; m < 2; ++m) {
        a1[m][0] = make_a_frag_g(dstp[m] + hlf * 8,      dstp[m] + 16 + hlf * 8);
        a1[m][1] = make_a_frag_g(dstp[m] + 32 + hlf * 8, dstp[m] + 48 + hlf * 8);
        a1[m][2] = make_a_frag_g(edgp[m] + hlf * 8,      edgp[m] + 16 + hlf * 8);
    }

    // ---- layer 1: K = 64..159 only; acc seeded with per-column src bias ----
#pragma unroll
    for (int n = 0; n < 8; ++n) {
        const float bv = biasp[n * 16 + lr];   // same for every row of the tile
        v8f cinit;
#pragma unroll
        for (int r = 0; r < 8; ++r) cinit[r] = bv;
        v8f acc0 = cinit, acc1 = cinit;
#pragma unroll
        for (int k = 0; k < 3; ++k) {
            const _Float16* bp = sWt + (n * 16 + lr) * D_IN + 64 + k * 32 + hlf * 8;
            v16h bw = load_frag_lds(bp, bp + 16);
            acc0 = WMMA_F16(a1[0][k], bw, acc0);
            acc1 = WMMA_F16(a1[1][k], bw, acc1);
        }
        relu_store_h(sH + (8 * hlf) * D_H + n * 16 + lr, acc0);
        relu_store_h(sH + (16 + 8 * hlf) * D_H + n * 16 + lr, acc1);
    }

    // ---- layer 2: A frags from sH (same-wave DS ops are in-order) ----
    v16h a2[2][4];
#pragma unroll
    for (int m = 0; m < 2; ++m)
#pragma unroll
        for (int c = 0; c < 4; ++c) {
            const _Float16* p0 = sH + (m * 16 + lr) * D_H + c * 32 + hlf * 8;
            a2[m][c] = load_frag_lds(p0, p0 + 16);
        }
#pragma unroll
    for (int n = 0; n < 8; ++n) {
        v8f acc0 = {}, acc1 = {};
#pragma unroll
        for (int k = 0; k < 4; ++k) {
            const _Float16* bp = sW1 + (n * 16 + lr) * D_H + k * 32 + hlf * 8;
            v16h bw = load_frag_lds(bp, bp + 16);
            acc0 = WMMA_F16(a2[0][k], bw, acc0);
            acc1 = WMMA_F16(a2[1][k], bw, acc1);
        }
        relu_store_h(sH + (8 * hlf) * D_H + n * 16 + lr, acc0);       // a2 already in regs
        relu_store_h(sH + (16 + 8 * hlf) * D_H + n * 16 + lr, acc1);
    }

    // ---- layer 3: 4 col-tiles, f32 result straight to global ----
    v16h a3[2][4];
#pragma unroll
    for (int m = 0; m < 2; ++m)
#pragma unroll
        for (int c = 0; c < 4; ++c) {
            const _Float16* p0 = sH + (m * 16 + lr) * D_H + c * 32 + hlf * 8;
            a3[m][c] = load_frag_lds(p0, p0 + 16);
        }
#pragma unroll
    for (int n = 0; n < 4; ++n) {
        v8f acc0 = {}, acc1 = {};
#pragma unroll
        for (int k = 0; k < 4; ++k) {
            const _Float16* bp = sW2 + (n * 16 + lr) * D_H + k * 32 + hlf * 8;
            v16h bw = load_frag_lds(bp, bp + 16);
            acc0 = WMMA_F16(a3[0][k], bw, acc0);
            acc1 = WMMA_F16(a3[1][k], bw, acc1);
        }
        float* op0 = out + (size_t)(row0 + 8 * hlf) * D_OUT + n * 16 + lr;
        float* op1 = out + (size_t)(row0 + 16 + 8 * hlf) * D_OUT + n * 16 + lr;
#pragma unroll
        for (int r = 0; r < 8; ++r) {
            op0[(size_t)r * D_OUT] = acc0[r];
            op1[(size_t)r * D_OUT] = acc1[r];
        }
    }
}

extern "C" void kernel_launch(void* const* d_in, const int* in_sizes, int n_in,
                              void* d_out, int out_size, void* d_ws, size_t ws_size,
                              hipStream_t stream) {
    const float* node = (const float*)d_in[0];  // [4, 384, 64]
    const float* edge = (const float*)d_in[1];  // [4, 147456, 32]
    const float* Wt   = (const float*)d_in[2];  // [128, 160]
    const float* W1   = (const float*)d_in[3];  // [128, 128]
    const float* W2   = (const float*)d_in[4];  // [64, 128]

    _Float16* wf16 = (_Float16*)d_ws;                           // 90112 B
    float* bias    = (float*)((char*)d_ws + (size_t)W_TOTAL * 2); // 1536*128 f32

    cvt_weights<<<(W_TOTAL + 255) / 256, 256, 0, stream>>>(Wt, W1, W2, wf16);
    src_bias_kernel<<<NB * NN, D_H, 0, stream>>>(node, Wt, bias);

    const int blocks = ROWS / 256;  // 2304, each block = 8 waves x 32 rows
    edge_mlp_kernel<<<blocks, 256, LDS_F16_TOTAL * 2, stream>>>(node, edge, wf16, bias,
                                                                (float*)d_out);
}
